// GCNII_68247030334296
// MI455X (gfx1250) — compile-verified, matrix operand order checked by reading, not run on hardware
//
#include <hip/hip_runtime.h>
#include <hip/hip_bf16.h>
#include <math.h>

#define NNODES 100000
#define NEDGES 3200000
#define NFEAT  512
#define NHID   256
#define NCLASS 40
#define NLAYERS 8
#define EPSV   1e-5f

typedef __attribute__((ext_vector_type(16))) __bf16 v16bf;
typedef __attribute__((ext_vector_type(8)))  float  v8f;

// ---------------------------------------------------------------------------
// small fp32 kernels for Newton-Schulz orthogonalization (tiny matrices)
// ---------------------------------------------------------------------------

// C[M,N] = alpha * A @ op(B) + beta * D   (op(B)=B^T if transB)
__global__ void k_gemm_f32(float* __restrict__ C, const float* __restrict__ A,
                           const float* __restrict__ B, const float* __restrict__ D,
                           int M, int N, int K, int transB, float alpha, float beta) {
    __shared__ float As[16][16];
    __shared__ float Bs[16][17];
    int tx = threadIdx.x, ty = threadIdx.y;
    int col = blockIdx.x * 16 + tx;
    int row = blockIdx.y * 16 + ty;
    float acc = 0.f;
    for (int k0 = 0; k0 < K; k0 += 16) {
        As[ty][tx] = A[(size_t)row * K + k0 + tx];
        if (transB)
            Bs[ty][tx] = B[(size_t)(blockIdx.x * 16 + tx) * K + k0 + ty];
        else
            Bs[ty][tx] = B[(size_t)(k0 + ty) * N + col];
        __syncthreads();
#pragma unroll
        for (int kk = 0; kk < 16; ++kk) acc += As[ty][kk] * Bs[kk][tx];
        __syncthreads();
    }
    float out = alpha * acc;
    if (D) out += beta * D[(size_t)row * N + col];
    C[(size_t)row * N + col] = out;
}

// Zc = W - rowmean(W); one block per row
__global__ void k_center(const float* __restrict__ W, float* __restrict__ Zc, int k) {
    __shared__ float red[256];
    int row = blockIdx.x, tid = threadIdx.x;
    float s = 0.f;
    for (int j = tid; j < k; j += 256) s += W[(size_t)row * k + j];
    red[tid] = s; __syncthreads();
    for (int o = 128; o > 0; o >>= 1) { if (tid < o) red[tid] += red[tid + o]; __syncthreads(); }
    float mean = red[0] / (float)k;
    for (int j = tid; j < k; j += 256) Zc[(size_t)row * k + j] = W[(size_t)row * k + j] - mean;
}

__global__ void k_eps_diag(float* S) { int i = threadIdx.x; S[i * 256 + i] += EPSV; }

__global__ void k_set_identity(float* B) {
    B[blockIdx.x * 256 + threadIdx.x] = (blockIdx.x == (int)threadIdx.x) ? 1.f : 0.f;
}

__global__ void k_sumsq(const float* __restrict__ X, int n, float* out) {
    __shared__ float red[256];
    float s = 0.f;
    for (int i = blockIdx.x * 256 + threadIdx.x; i < n; i += gridDim.x * 256) s += X[i] * X[i];
    red[threadIdx.x] = s; __syncthreads();
    for (int o = 128; o > 0; o >>= 1) { if ((int)threadIdx.x < o) red[threadIdx.x] += red[threadIdx.x + o]; __syncthreads(); }
    if (threadIdx.x == 0) atomicAdd(out, red[0]);
}

// X *= 1/sqrt(sumsq)  (i.e. divide by frobenius norm)
__global__ void k_scale_rsqrt(float* X, int n, const float* sumsq) {
    float f = rsqrtf(*sumsq);
    for (int i = blockIdx.x * 256 + threadIdx.x; i < n; i += gridDim.x * 256) X[i] *= f;
}

// X *= 1/sqrt(norm) = sumsq^(-1/4)
__global__ void k_scale_qrt(float* X, int n, const float* sumsq) {
    float f = rsqrtf(sqrtf(*sumsq));
    for (int i = blockIdx.x * 256 + threadIdx.x; i < n; i += gridDim.x * 256) X[i] *= f;
}

// we = 0.5*convW[l] + 0.5*I
__global__ void k_make_we(float* we, const float* __restrict__ convW, int l) {
    int i = blockIdx.x, j = threadIdx.x;
    float v = 0.5f * convW[(size_t)l * NHID * NHID + i * NHID + j];
    if (i == j) v += 0.5f;
    we[i * NHID + j] = v;
}

// tT[n][k] = bf16(t[k][n])   (256x256)
__global__ void k_transpose_bf16(__hip_bfloat16* __restrict__ dst, const float* __restrict__ src) {
    int n = blockIdx.x * 16 + threadIdx.x;
    int k = blockIdx.y * 16 + threadIdx.y;
    dst[(size_t)n * 256 + k] = __float2bfloat16(src[(size_t)k * 256 + n]);
}

__global__ void k_cast_bf16(__hip_bfloat16* __restrict__ dst, const float* __restrict__ src, size_t n) {
    size_t i = (size_t)blockIdx.x * 256 + threadIdx.x;
    if (i < n) dst[i] = __float2bfloat16(src[i]);
}

// ---------------------------------------------------------------------------
// CSR build (by destination) -- once per call
// ---------------------------------------------------------------------------
__global__ void k_deg_count(const int* __restrict__ dst, unsigned* deg, int E) {
    int e = blockIdx.x * 256 + threadIdx.x;
    if (e < E) atomicAdd(&deg[dst[e]], 1u);
}

__global__ void k_scan_block(const unsigned* __restrict__ deg, unsigned* incl, unsigned* bsum, int n) {
    __shared__ unsigned s[256];
    int i = blockIdx.x * 256 + threadIdx.x, tid = threadIdx.x;
    unsigned v = (i < n) ? deg[i] : 0u;
    s[tid] = v; __syncthreads();
    for (int o = 1; o < 256; o <<= 1) {
        unsigned t = (tid >= o) ? s[tid - o] : 0u; __syncthreads();
        s[tid] += t; __syncthreads();
    }
    if (i < n) incl[i] = s[tid];
    if (tid == 255) bsum[blockIdx.x] = s[255];
}

__global__ void k_scan_sums(unsigned* sums, int nb) {
    __shared__ unsigned s[512];
    int tid = threadIdx.x;
    unsigned v = (tid < nb) ? sums[tid] : 0u;
    s[tid] = v; __syncthreads();
    for (int o = 1; o < 512; o <<= 1) {
        unsigned t = (tid >= o) ? s[tid - o] : 0u; __syncthreads();
        s[tid] += t; __syncthreads();
    }
    if (tid < nb) sums[tid] = s[tid] - v;   // exclusive block offset
}

__global__ void k_finalize_rowptr(const unsigned* __restrict__ deg, const unsigned* __restrict__ incl,
                                  const unsigned* __restrict__ bsum, unsigned* rowptr,
                                  unsigned* cursor, int n, int E) {
    int i = blockIdx.x * 256 + threadIdx.x;
    if (i < n) {
        unsigned r = bsum[i / 256] + incl[i] - deg[i];
        rowptr[i] = r; cursor[i] = r;
    }
    if (i == 0) rowptr[n] = (unsigned)E;
}

__global__ void k_scatter(const int* __restrict__ src, const int* __restrict__ dst,
                          const float* __restrict__ w, unsigned* cursor,
                          int* __restrict__ es, float* __restrict__ ew, int E) {
    int e = blockIdx.x * 256 + threadIdx.x;
    if (e < E) {
        unsigned pos = atomicAdd(&cursor[dst[e]], 1u);
        es[pos] = src[e];
        ew[pos] = w[e];
    }
}

// ---------------------------------------------------------------------------
// SpMM (gather CSR) fused with support = 0.9*agg + 0.1*h0, output bf16
// one wave per node; each lane owns 8 contiguous features (one b128 / edge)
// ---------------------------------------------------------------------------
__global__ void k_spmm(const unsigned* __restrict__ rowptr, const int* __restrict__ es,
                       const float* __restrict__ ew, const __hip_bfloat16* __restrict__ h,
                       const __hip_bfloat16* __restrict__ h0, __hip_bfloat16* __restrict__ sup) {
    int node = blockIdx.x * 8 + (threadIdx.x >> 5);
    if (node >= NNODES) return;
    int lane = threadIdx.x & 31;
    int fb = lane * 8;
    float acc[8];
#pragma unroll
    for (int i = 0; i < 8; ++i) acc[i] = 0.f;
    unsigned beg = rowptr[node], end = rowptr[node + 1];
    for (unsigned e = beg; e < end; ++e) {
        int s = es[e];
        float w = ew[e];
        if (e + 1 < end) __builtin_prefetch(h + (size_t)es[e + 1] * NHID + fb, 0, 1);
        union { uint4 u; __hip_bfloat16 b[8]; } r;
        r.u = *reinterpret_cast<const uint4*>(h + (size_t)s * NHID + fb);
#pragma unroll
        for (int i = 0; i < 8; ++i) acc[i] += w * __bfloat162float(r.b[i]);
    }
    union { uint4 u; __hip_bfloat16 b[8]; } r0, o;
    r0.u = *reinterpret_cast<const uint4*>(h0 + (size_t)node * NHID + fb);
#pragma unroll
    for (int i = 0; i < 8; ++i)
        o.b[i] = __float2bfloat16(0.9f * acc[i] + 0.1f * __bfloat162float(r0.b[i]));
    *reinterpret_cast<uint4*>(sup + (size_t)node * NHID + fb) = o.u;
}

// ---------------------------------------------------------------------------
// bf16 WMMA GEMM: C[nrows,256] = A[nrows,K] @ B, with BT[n][k] = B[k][n] given
// row-major. Block stages its 64-column B strip in LDS (padded rows), then
// each wave computes a 32x64 strip (2 M-tiles x 4 N-tiles => 8 accumulators,
// B fragments from LDS reused across both M-tiles). nrows must be a multiple
// of 32 so both M-tiles of a wave share one validity guard (branch-free loop).
// Epilogue:
//   mode 0: relu(acc + bias[col])                     -> out0, out1 (h and h0)
//   mode 1: relu(theta*acc + (1-theta)*sup[row][col]) -> out0 (h)
// A-fragment layout per ISA 16-bit A 16x32 table (ksel = 8*(lane>=16)).
// ---------------------------------------------------------------------------
__device__ __forceinline__ v16bf load_frag_g(const __hip_bfloat16* p, int ksel) {
    union { uint4 u[2]; v16bf v; } f;
    f.u[0] = *reinterpret_cast<const uint4*>(p + ksel);
    f.u[1] = *reinterpret_cast<const uint4*>(p + ksel + 16);
    return f.v;
}

__global__ void k_wmma_gemm(const __hip_bfloat16* __restrict__ A,
                            const __hip_bfloat16* __restrict__ BT,
                            __hip_bfloat16* __restrict__ out0,
                            __hip_bfloat16* __restrict__ out1,
                            const float* __restrict__ bias,
                            const __hip_bfloat16* __restrict__ sup,
                            int nrows, int K, int mode, float theta) {
    extern __shared__ __hip_bfloat16 ldsB[];   // [64][K+8] bf16 (padded rows)
    const int ldb = K + 8;
    const int lane  = threadIdx.x & 31;
    const int wv    = threadIdx.x >> 5;
    const int l15   = lane & 15;
    const int ksel  = (lane >> 4) * 8;
    const int rbase = 8 * (lane >> 4);
    const int n0    = blockIdx.y * 64;

    // cooperative stage of B strip (rows n0..n0+63 of BT) into LDS
    const int kchunks = K / 8;
    for (int idx = threadIdx.x; idx < 64 * kchunks; idx += 256) {
        int row = idx / kchunks;
        int kc  = (idx - row * kchunks) * 8;
        *reinterpret_cast<uint4*>(&ldsB[row * ldb + kc]) =
            *reinterpret_cast<const uint4*>(BT + (size_t)(n0 + row) * K + kc);
    }
    __syncthreads();

    const int mt0 = (blockIdx.x * 8 + wv) * 2;       // first of two M-tiles
    if (mt0 * 16 + 32 > nrows) return;               // after the barrier: safe

    v8f acc[2][4];
#pragma unroll
    for (int m = 0; m < 2; ++m)
#pragma unroll
        for (int j = 0; j < 4; ++j)
            acc[m][j] = (v8f){0.f, 0.f, 0.f, 0.f, 0.f, 0.f, 0.f, 0.f};

    const __hip_bfloat16* arow0 = A + (size_t)(mt0 * 16 + l15) * K;
    const __hip_bfloat16* arow1 = A + (size_t)((mt0 + 1) * 16 + l15) * K;

    for (int kb = 0; kb < K; kb += 32) {
        v16bf a0 = load_frag_g(arow0 + kb, ksel);
        v16bf a1 = load_frag_g(arow1 + kb, ksel);
#pragma unroll
        for (int j = 0; j < 4; ++j) {
            union { uint4 u[2]; v16bf v; } bf;
            const __hip_bfloat16* bp = &ldsB[(j * 16 + l15) * ldb + kb + ksel];
            bf.u[0] = *reinterpret_cast<const uint4*>(bp);
            bf.u[1] = *reinterpret_cast<const uint4*>(bp + 16);
            acc[0][j] = __builtin_amdgcn_wmma_f32_16x16x32_bf16(
                false, a0, false, bf.v, (short)0, acc[0][j], false, false);
            acc[1][j] = __builtin_amdgcn_wmma_f32_16x16x32_bf16(
                false, a1, false, bf.v, (short)0, acc[1][j], false, false);
        }
    }

#pragma unroll
    for (int m = 0; m < 2; ++m) {
        int mt = mt0 + m;
#pragma unroll
        for (int j = 0; j < 4; ++j) {
            int col = n0 + j * 16 + l15;
#pragma unroll
            for (int r = 0; r < 8; ++r) {
                size_t idx = (size_t)(mt * 16 + rbase + r) * NHID + col;
                float v = acc[m][j][r];
                if (mode == 0) {
                    v += bias[col];
                    v = v > 0.f ? v : 0.f;
                    __hip_bfloat16 bv = __float2bfloat16(v);
                    out0[idx] = bv;
                    out1[idx] = bv;
                } else {
                    float s = __bfloat162float(sup[idx]);
                    v = theta * v + (1.f - theta) * s;
                    v = v > 0.f ? v : 0.f;
                    out0[idx] = __float2bfloat16(v);
                }
            }
        }
    }
}

// ---------------------------------------------------------------------------
// classifier: logits = h @ W1^T + b1 ; log_softmax. One block (64 thr) / node.
// ---------------------------------------------------------------------------
__global__ void k_classifier(const __hip_bfloat16* __restrict__ h,
                             const float* __restrict__ W1, const float* __restrict__ b1,
                             float* __restrict__ out) {
    __shared__ float logits[NCLASS];
    __shared__ float red[2];
    int node = blockIdx.x;
    int c = threadIdx.x;
    if (c < NCLASS) {
        float acc = b1[c];
        const __hip_bfloat16* hrow = h + (size_t)node * NHID;
        const float* wrow = W1 + (size_t)c * NHID;
        for (int k = 0; k < NHID; k += 8) {
            union { uint4 u; __hip_bfloat16 b[8]; } r;
            r.u = *reinterpret_cast<const uint4*>(hrow + k);
#pragma unroll
            for (int i = 0; i < 8; ++i) acc += __bfloat162float(r.b[i]) * wrow[k + i];
        }
        logits[c] = acc;
    }
    __syncthreads();
    if (threadIdx.x == 0) {
        float mx = -1e30f;
        for (int i = 0; i < NCLASS; ++i) mx = fmaxf(mx, logits[i]);
        float s = 0.f;
        for (int i = 0; i < NCLASS; ++i) s += expf(logits[i] - mx);
        red[0] = mx; red[1] = logf(s);
    }
    __syncthreads();
    if (c < NCLASS) out[(size_t)node * NCLASS + c] = logits[c] - red[0] - red[1];
}

// ---------------------------------------------------------------------------
// host orchestration
// ---------------------------------------------------------------------------
extern "C" void kernel_launch(void* const* d_in, const int* in_sizes, int n_in,
                              void* d_out, int out_size, void* d_ws, size_t ws_size,
                              hipStream_t stream) {
    const float* x     = (const float*)d_in[0];
    const int*   esrc  = (const int*)d_in[1];
    const int*   edst  = (const int*)d_in[2];
    const float* ewt   = (const float*)d_in[3];
    const float* W0    = (const float*)d_in[4];
    const float* b0    = (const float*)d_in[5];
    const float* convW = (const float*)d_in[6];
    const float* W1    = (const float*)d_in[7];
    const float* b1    = (const float*)d_in[8];

    char* base = (char*)d_ws;
    size_t off = 0;
    auto alloc = [&](size_t bytes) -> void* {
        void* p = base + off;
        off += (bytes + 255) & ~(size_t)255;
        return p;
    };

    __hip_bfloat16* xbf   = (__hip_bfloat16*)alloc((size_t)NNODES * NFEAT * 2);
    __hip_bfloat16* hbf   = (__hip_bfloat16*)alloc((size_t)NNODES * NHID * 2);
    __hip_bfloat16* h0bf  = (__hip_bfloat16*)alloc((size_t)NNODES * NHID * 2);
    __hip_bfloat16* supbf = (__hip_bfloat16*)alloc((size_t)NNODES * NHID * 2);
    float*          W0o   = (float*)alloc((size_t)NHID * NFEAT * 4);
    __hip_bfloat16* W0obf = (__hip_bfloat16*)alloc((size_t)NHID * NFEAT * 2);
    __hip_bfloat16* tTbf  = (__hip_bfloat16*)alloc((size_t)NLAYERS * NHID * NHID * 2);
    float*          tF32  = (float*)alloc((size_t)NHID * NHID * 4);
    float*          weB   = (float*)alloc((size_t)NHID * NHID * 4);
    float*          Zc    = (float*)alloc((size_t)NHID * NFEAT * 4);
    float*          S     = (float*)alloc((size_t)NHID * NHID * 4);
    float*          B2    = (float*)alloc((size_t)NHID * NHID * 4);
    float*          B3    = (float*)alloc((size_t)NHID * NHID * 4);
    float*          Bp    = (float*)alloc((size_t)NHID * NHID * 4);
    float*          Bq    = (float*)alloc((size_t)NHID * NHID * 4);
    float*          scal  = (float*)alloc(256);
    unsigned*       deg   = (unsigned*)alloc((size_t)NNODES * 4);
    unsigned*       incl  = (unsigned*)alloc((size_t)NNODES * 4);
    unsigned*       curs  = (unsigned*)alloc((size_t)NNODES * 4);
    unsigned*       rowp  = (unsigned*)alloc((size_t)(NNODES + 1) * 4);
    unsigned*       bsum  = (unsigned*)alloc(512 * 4);
    int*            es    = (int*)alloc((size_t)NEDGES * 4);
    float*          ew    = (float*)alloc((size_t)NEDGES * 4);

    auto gemm = [&](float* C, const float* A, const float* B, const float* D,
                    int M, int N, int K, int transB, float alpha, float beta) {
        dim3 g(N / 16, M / 16), b(16, 16);
        k_gemm_f32<<<g, b, 0, stream>>>(C, A, B, D, M, N, K, transB, alpha, beta);
    };

    // Newton-Schulz orthogonalization: Wout = ortho(Win), Win is 256 x k
    auto ortho = [&](const float* Win, float* Wout, int k, int T) {
        k_center<<<NHID, 256, 0, stream>>>(Win, Zc, k);
        gemm(S, Zc, Zc, nullptr, NHID, NHID, k, 1, 1.f, 0.f);          // S = Zc @ Zc^T
        k_eps_diag<<<1, NHID, 0, stream>>>(S);
        hipMemsetAsync(scal, 0, 4, stream);
        k_sumsq<<<64, 256, 0, stream>>>(S, NHID * NHID, scal);
        k_scale_rsqrt<<<64, 256, 0, stream>>>(S, NHID * NHID, scal);   // S /= ||S||_F
        k_set_identity<<<NHID, NHID, 0, stream>>>(Bp);
        float* bc = Bp; float* bn = Bq;
        for (int t = 0; t < T; ++t) {
            gemm(B2, bc, bc, nullptr, NHID, NHID, NHID, 0, 1.f, 0.f);
            gemm(B3, B2, bc, nullptr, NHID, NHID, NHID, 0, 1.f, 0.f);
            gemm(bn, B3, S, bc, NHID, NHID, NHID, 0, -0.5f, 1.5f);     // 1.5B - 0.5 B^3 S
            float* tmp = bc; bc = bn; bn = tmp;
        }
        gemm(Wout, bc, Zc, nullptr, NHID, k, NHID, 0, 1.f, 0.f);
        k_scale_qrt<<<256, 256, 0, stream>>>(Wout, NHID * k, scal);    // / sqrt(||S||)
    };

    // 1. cast x to bf16
    {
        size_t n = (size_t)NNODES * NFEAT;
        k_cast_bf16<<<(unsigned)((n + 255) / 256), 256, 0, stream>>>(xbf, x, n);
    }

    // 2. CSR by destination
    hipMemsetAsync(deg, 0, (size_t)NNODES * 4, stream);
    k_deg_count<<<(NEDGES + 255) / 256, 256, 0, stream>>>(edst, deg, NEDGES);
    int nb = (NNODES + 255) / 256;
    k_scan_block<<<nb, 256, 0, stream>>>(deg, incl, bsum, NNODES);
    k_scan_sums<<<1, 512, 0, stream>>>(bsum, nb);
    k_finalize_rowptr<<<nb, 256, 0, stream>>>(deg, incl, bsum, rowp, curs, NNODES, NEDGES);
    k_scatter<<<(NEDGES + 255) / 256, 256, 0, stream>>>(esrc, edst, ewt, curs, es, ew, NEDGES);

    // 3. orthogonalize weights
    ortho(W0, W0o, NFEAT, 5);
    {
        size_t n = (size_t)NHID * NFEAT;
        k_cast_bf16<<<(unsigned)((n + 255) / 256), 256, 0, stream>>>(W0obf, W0o, n);
    }
    for (int l = 0; l < NLAYERS; ++l) {
        k_make_we<<<NHID, NHID, 0, stream>>>(weB, convW, l);
        ortho(weB, tF32, NHID, 5);
        dim3 g(16, 16), b(16, 16);
        k_transpose_bf16<<<g, b, 0, stream>>>(tTbf + (size_t)l * NHID * NHID, tF32);
    }

    const int mtiles = NNODES / 16;                 // 6250
    const int gx = (mtiles + 15) / 16;              // 16 M-tiles per block

    // 4. FC1: h = h0 = relu(x @ W0o^T + b0)   [bf16 WMMA, B strip in LDS]
    {
        dim3 g(gx, NHID / 64);
        size_t shmem = (size_t)64 * (NFEAT + 8) * 2;
        k_wmma_gemm<<<g, 256, shmem, stream>>>(xbf, W0obf, hbf, h0bf, b0, nullptr,
                                               NNODES, NFEAT, 0, 0.f);
    }

    // 5. GCNII layers
    for (int l = 1; l <= NLAYERS; ++l) {
        k_spmm<<<(NNODES + 7) / 8, 256, 0, stream>>>(rowp, es, ew, hbf, h0bf, supbf);
        float theta = logf(0.5f / (float)l + 1.f);
        dim3 g(gx, NHID / 64);
        size_t shmem = (size_t)64 * (NHID + 8) * 2;
        k_wmma_gemm<<<g, 256, shmem, stream>>>(supbf, tTbf + (size_t)(l - 1) * NHID * NHID,
                                               hbf, nullptr, nullptr, supbf,
                                               NNODES, NHID, 1, theta);
    }

    // 6. classifier + log_softmax
    k_classifier<<<NNODES, 64, 0, stream>>>(hbf, W1, b1, (float*)d_out);
}
